// Convolution_90357521973616
// MI455X (gfx1250) — matrix operand order repaired and checked by reading, not compiled, not run on hardware
//
#include <hip/hip_runtime.h>

typedef __attribute__((ext_vector_type(16))) _Float16 v16h;
typedef __attribute__((ext_vector_type(8)))  _Float16 v8h;
typedef __attribute__((ext_vector_type(8)))  float    v8f;

#define MULQ   16
#define DIMEMB 64
#define HID    256
#define WN     1024         // 4*16*16 output cols of W2
#define EPB    64           // edges per block (4 waves x 16)
#define NWAVE  4
#define PITCH  264          // halves per transposed row in stage buffer (bank-conflict pad)
#define BUFH   (16 * PITCH) // halves per stage buffer
#define BUFB   (BUFH * 2)   // bytes per stage buffer (8448)

typedef __attribute__((address_space(3))) char lds_char;

__device__ __forceinline__ float silu_f(float x) {
    // x * sigmoid(x) with fast v_rcp_f32 (no IEEE divide expansion)
    return x * __builtin_amdgcn_rcpf(1.0f + __expf(-x));
}

// one 16B async copy global->LDS (GVS mode: 64-bit sgpr base + 32-bit vgpr offset)
__device__ __forceinline__ void async_cp16(lds_char* lds, unsigned goff,
                                           unsigned long long gbase) {
    asm volatile("global_load_async_to_lds_b128 %0, %1, %2"
                 :: "v"(lds), "v"(goff), "s"(gbase)
                 : "memory");
}
__device__ __forceinline__ void wait_async_le4() {
    asm volatile("s_wait_asynccnt 0x4" ::: "memory");
}
__device__ __forceinline__ void wait_async_0() {
    asm volatile("s_wait_asynccnt 0x0" ::: "memory");
}

// B fragment (32x16 f16) from transposed LDS stage: lane n=l%16 reads 16
// consecutive K at k0 (+16 for upper half-wave). Two 16B ds_load_b128.
__device__ __forceinline__ v16h load_bfrag(const _Float16* stg, int n, int k0) {
    const v8h* p = (const v8h*)(stg + n * PITCH + k0);
    v8h lo = p[0];
    v8h hi = p[1];
    v16h b;
#pragma unroll
    for (int i = 0; i < 8; ++i) { b[i] = lo[i]; b[i + 8] = hi[i]; }
    return b;
}

// ---- pre-pass: convert W1/W2 to f16, transposed, tile-major -----------------
// w1t: [16 tiles][16 n][64 k]   (32 KB)
// w2t: [64 tiles][16 n][256 k]  (512 KB)
__global__ __launch_bounds__(256) void convert_weights(
    const float* __restrict__ W1, const float* __restrict__ W2,
    _Float16* __restrict__ w1t, _Float16* __restrict__ w2t)
{
    const int i = blockIdx.x * 256 + threadIdx.x;
    if (i < 16 * 16 * 64) {
        int t = i >> 10, r = i & 1023, n = r >> 6, k = r & 63;
        w1t[i] = (_Float16)W1[k * HID + t * 16 + n];
    }
    for (int o = i; o < 64 * 16 * 256; o += gridDim.x * 256) {
        int j = o >> 12, r = o & 4095, n = r >> 8, k = r & 255;
        w2t[o] = (_Float16)W2[(size_t)k * WN + j * 16 + n];
    }
}

template <bool ASY>
__global__ __launch_bounds__(128) void edge_conv_fused(
    const float* __restrict__ nodef,  // (N,64)
    const int*   __restrict__ esrc,   // (E)
    const int*   __restrict__ edst,   // (E)
    const float* __restrict__ eattr,  // (E,4)
    const float* __restrict__ eemb,   // (E,64)
    const float* __restrict__ W1,     // (64,256)   (fallback path)
    const float* __restrict__ b1,     // (256)
    const float* __restrict__ W2,     // (256,1024) (fallback path)
    const float* __restrict__ b2,     // (1024)
    const _Float16* __restrict__ w1t, // pre-converted (ASY path)
    const _Float16* __restrict__ w2t, // pre-converted (ASY path)
    float*       __restrict__ out)    // (N,64), pre-zeroed, atomic accumulate
{
    __shared__ _Float16 stage[3 * BUFH];        // 25344 B: triple-buffered weight tiles
    __shared__ float    coef[NWAVE * 128 * 16]; // 32768 B: compact epilogue coefficients
    __shared__ int      dsts[NWAVE * 16];

    const int tid    = threadIdx.x;
    const int lane   = tid & 31;
    const int wave   = tid >> 5;
    const int l16    = lane & 15;
    const int lhi    = lane >> 4;      // 0 / 1
    const int rowoff = lhi * 8;        // D-fragment row offset for this lane half
    const int eb     = blockIdx.x * EPB + wave * MULQ;  // first edge of this wave

    float*    coefW = &coef[wave * (128 * 16)];
    _Float16* gbuf  = (_Float16*)coefW;  // 16x32 f16 scratch, reused before coef written

    // async staging geometry: thread -> (row n, 16B sub-chunk)
    const int cn  = tid >> 3;          // 0..15
    const int cs  = tid & 7;           // 0..7
    lds_char* lds0 = (lds_char*)&stage[0];
    const unsigned long long w1base = (unsigned long long)(uintptr_t)w1t;
    const unsigned long long w2base = (unsigned long long)(uintptr_t)w2t;
    const unsigned ldsoff_w2 = (unsigned)(cn * (PITCH * 2) + cs * 64);
    const unsigned goff_w2   = (unsigned)(cn * 512 + cs * 64);

    // ---------------- Phase 0: edge-embedding A fragments (16x64 f16) --------
    v16h aemb[2];
    {
        const float* rp = eemb + (size_t)(eb + l16) * DIMEMB;
#pragma unroll
        for (int kb = 0; kb < 2; ++kb) {
            const int c1 = kb * 32 + lhi * 16;  // K chunk {c1..c1+15}
            float4 f0 = *(const float4*)(rp + c1);
            float4 f1 = *(const float4*)(rp + c1 + 4);
            float4 f2 = *(const float4*)(rp + c1 + 8);
            float4 f3 = *(const float4*)(rp + c1 + 12);
            v16h a;
            a[0]  = (_Float16)f0.x; a[1]  = (_Float16)f0.y; a[2]  = (_Float16)f0.z; a[3]  = (_Float16)f0.w;
            a[4]  = (_Float16)f1.x; a[5]  = (_Float16)f1.y; a[6]  = (_Float16)f1.z; a[7]  = (_Float16)f1.w;
            a[8]  = (_Float16)f2.x; a[9]  = (_Float16)f2.y; a[10] = (_Float16)f2.z; a[11] = (_Float16)f2.w;
            a[12] = (_Float16)f3.x; a[13] = (_Float16)f3.y; a[14] = (_Float16)f3.z; a[15] = (_Float16)f3.w;
            aemb[kb] = a;
        }
    }

    // ---------------- Phase 1: G = silu(emb @ W1 + b1), kept as A fragments ----
    v16h ga[8];  // 16 edges x 256 hidden as 8 A fragments, live across the W2 loop
#pragma unroll
    for (int kb = 0; kb < 8; ++kb) {
#pragma unroll
        for (int tt = 0; tt < 2; ++tt) {
            const int t = kb * 2 + tt;  // hidden col tile
            __syncthreads();
            if (ASY) {
                // one 16B async chunk per thread: 16 rows x 128 B into buffer 0
                async_cp16(lds0 + cn * (PITCH * 2) + cs * 16,
                           (unsigned)(t * 2048 + cn * 128 + cs * 16), w1base);
                wait_async_0();
            } else {
#pragma unroll
                for (int i = 0; i < 8; ++i) {
                    int idx = tid + i * 128;
                    int n = idx & 15, k = idx >> 4;
                    stage[n * PITCH + k] = (_Float16)W1[k * HID + t * 16 + n];
                }
            }
            __syncthreads();
            float b1v = b1[t * 16 + l16];
            v8f acc;
#pragma unroll
            for (int r = 0; r < 8; ++r) acc[r] = b1v;
            v16h bf0 = load_bfrag(stage, l16, 0 + 16 * lhi);
            acc = __builtin_amdgcn_wmma_f32_16x16x32_f16(false, aemb[0], false, bf0,
                                                         (short)0, acc, false, false);
            v16h bf1 = load_bfrag(stage, l16, 32 + 16 * lhi);
            acc = __builtin_amdgcn_wmma_f32_16x16x32_f16(false, aemb[1], false, bf1,
                                                         (short)0, acc, false, false);
#pragma unroll
            for (int r = 0; r < 8; ++r) {
                float g = silu_f(acc[r]);
                gbuf[(r + rowoff) * 32 + tt * 16 + l16] = (_Float16)g;
            }
        }
        {   // reload this 32-wide K block as an A fragment
            const _Float16* p = gbuf + (size_t)l16 * 32 + lhi * 16;
            v8h lo = *(const v8h*)p;
            v8h hi = *(const v8h*)(p + 8);
            v16h a;
#pragma unroll
            for (int i = 0; i < 8; ++i) { a[i] = lo[i]; a[i + 8] = hi[i]; }
            ga[kb] = a;
        }
    }

    // ---------------- Phase 2: per-edge epilogue coefficients into LDS --------
    // compact layout: rows [0,16)=ss | [16,64)=vs(u,m) | [64,112)=sv(u,m) | [112,128)=vv
    __syncthreads();
    if (lane < 16) {
        const int e = eb + lane;
        dsts[wave * 16 + lane] = edst[e];
        const int    srow = esrc[e];
        const float4 ea   = *(const float4*)(eattr + (size_t)e * 4);
        const float  s2 = ea.x, v2x = ea.y, v2y = ea.z, v2z = ea.w;
        const float* nf = nodef + (size_t)srow * 64;
        float s1[16];
#pragma unroll
        for (int i = 0; i < 4; ++i) {
            float4 f = *(const float4*)(nf + i * 4);
            s1[i * 4 + 0] = f.x; s1[i * 4 + 1] = f.y; s1[i * 4 + 2] = f.z; s1[i * 4 + 3] = f.w;
        }
        const float KS = 0.25f * 0.17677669529663687f;   // (1/sqrt(nbr)) * ALPHA
        const float KV = KS * 0.5773502691896258f;       // * C3
#pragma unroll
        for (int u = 0; u < 16; ++u) {
            float va = nf[16 + u * 3 + 0];
            float vb = nf[16 + u * 3 + 1];
            float vc = nf[16 + u * 3 + 2];
            coefW[(u) * 16 + lane]              = KS * s1[u] * s2;
            coefW[(16 + u * 3 + 0) * 16 + lane] = KV * va * s2;
            coefW[(16 + u * 3 + 1) * 16 + lane] = KV * vb * s2;
            coefW[(16 + u * 3 + 2) * 16 + lane] = KV * vc * s2;
            coefW[(64 + u * 3 + 0) * 16 + lane] = KV * s1[u] * v2x;
            coefW[(64 + u * 3 + 1) * 16 + lane] = KV * s1[u] * v2y;
            coefW[(64 + u * 3 + 2) * 16 + lane] = KV * s1[u] * v2z;
            coefW[(112 + u) * 16 + lane]        = KV * (va * v2x + vb * v2y + vc * v2z);
        }
    }
    __syncthreads();

    // ---------------- Phase 3: stream W2 16-col tiles, triple-buffered --------
    v8f souts = {};
    v8f outv0 = {}, outv1 = {}, outv2 = {};

    if (ASY) {  // prologue: start tile 0 into buffer 0
#pragma unroll
        for (int c = 0; c < 4; ++c)
            async_cp16(lds0 + 0 * BUFB + ldsoff_w2 + c * 16,
                       0u * 8192u + goff_w2 + c * 16, w2base);
    }

#pragma unroll 1
    for (int j = 0; j < 64; ++j) {
        const int b = ASY ? (j % 3) : 0;
        if (ASY) {
            if (j + 1 < 64) {   // overlap: fetch next tile while computing this one
                const int bn = (j + 1) % 3;
#pragma unroll
                for (int c = 0; c < 4; ++c)
                    async_cp16(lds0 + bn * BUFB + ldsoff_w2 + c * 16,
                               (unsigned)((j + 1) * 8192) + goff_w2 + c * 16, w2base);
                wait_async_le4();   // tile j's copies (older, in-order) are done
            } else {
                wait_async_0();
            }
            __syncthreads();
        } else {
            __syncthreads();
#pragma unroll
            for (int i = 0; i < 32; ++i) {
                int idx = tid + i * 128;
                int n = idx & 15, k = idx >> 4;
                stage[n * PITCH + k] = (_Float16)W2[(size_t)k * WN + j * 16 + n];
            }
            __syncthreads();
        }
        const _Float16* stg = &stage[b * BUFH];

        float b2v = b2[j * 16 + l16];
        v8f acc0, acc1 = {};
#pragma unroll
        for (int r = 0; r < 8; ++r) acc0[r] = b2v;
        // two independent WMMA chains to hide D->C hazard latency
#pragma unroll
        for (int kk = 0; kk < 4; ++kk) {
            v16h bfe = load_bfrag(stg, l16, (2 * kk) * 32 + 16 * lhi);
            acc0 = __builtin_amdgcn_wmma_f32_16x16x32_f16(false, ga[2 * kk], false, bfe,
                                                          (short)0, acc0, false, false);
            v16h bfo = load_bfrag(stg, l16, (2 * kk + 1) * 32 + 16 * lhi);
            acc1 = __builtin_amdgcn_wmma_f32_16x16x32_f16(false, ga[2 * kk + 1], false, bfo,
                                                          (short)0, acc1, false, false);
        }
        v8f acc = acc0 + acc1;

        // epilogue: per-row scalar FMA with precomputed coefficients
        const int base = (j < 16) ? j
                        : (j < 32) ? 16 + (j - 16) * 3
                        : (j < 48) ? 64 + (j - 32) * 3
                                   : 112 + (j - 48);
        const float* cb = coefW + base * 16 + rowoff;   // 32B-aligned
        if (j < 16 || j >= 48) {
            v8f c0 = *(const v8f*)cb;
            souts += c0 * acc;
        } else {
            v8f c0 = *(const v8f*)(cb);
            v8f c1 = *(const v8f*)(cb + 16);
            v8f c2 = *(const v8f*)(cb + 32);
            outv0 += c0 * acc;
            outv1 += c1 * acc;
            outv2 += c2 * acc;
        }
    }

    // ---------------- Phase 4: scatter-add into out[dst] ----------------------
#pragma unroll
    for (int r = 0; r < 8; ++r) {
        const int d = dsts[wave * 16 + rowoff + r];
        float* op = out + (size_t)d * 64;
        atomicAdd(op + l16, souts[r]);          // out_s channel l16
        float* opv = op + 16 + l16 * 3;         // out_v: 16 + v*3 + m
        atomicAdd(opv + 0, outv0[r]);
        atomicAdd(opv + 1, outv1[r]);
        atomicAdd(opv + 2, outv2[r]);
    }
}

extern "C" void kernel_launch(void* const* d_in, const int* in_sizes, int n_in,
                              void* d_out, int out_size, void* d_ws, size_t ws_size,
                              hipStream_t stream) {
    (void)n_in;
    const float* nodef = (const float*)d_in[0];
    const int*   esrc  = (const int*)  d_in[1];
    const int*   edst  = (const int*)  d_in[2];
    const float* eattr = (const float*)d_in[3];
    const float* eemb  = (const float*)d_in[4];
    const float* W1    = (const float*)d_in[5];
    const float* b1    = (const float*)d_in[6];
    const float* W2    = (const float*)d_in[7];
    const float* b2    = (const float*)d_in[8];
    float* out = (float*)d_out;

    const int n_edges = in_sizes[1];          // 160000, divisible by EPB
    const int blocks  = n_edges / EPB;

    hipMemsetAsync(d_out, 0, (size_t)out_size * sizeof(float), stream);

    const size_t w1t_bytes = (size_t)16 * 16 * 64 * 2;    // 32 KB
    const size_t w2t_bytes = (size_t)64 * 16 * 256 * 2;   // 512 KB
    if (d_ws && ws_size >= w1t_bytes + w2t_bytes) {
        _Float16* w1t = (_Float16*)d_ws;
        _Float16* w2t = (_Float16*)((char*)d_ws + w1t_bytes);
        convert_weights<<<1024, 256, 0, stream>>>(W1, W2, w1t, w2t);
        edge_conv_fused<true><<<blocks, 128, 0, stream>>>(
            nodef, esrc, edst, eattr, eemb, W1, b1, W2, b2, w1t, w2t, out);
    } else {
        edge_conv_fused<false><<<blocks, 128, 0, stream>>>(
            nodef, esrc, edst, eattr, eemb, W1, b1, W2, b2, nullptr, nullptr, out);
    }
}
